// GrangerCausalityNetwork_50259707297822
// MI455X (gfx1250) — matrix operand order repaired and checked by reading, not compiled, not run on hardware
//
#include <hip/hip_runtime.h>
#include <hip/hip_bf16.h>

#define NV 32
#define LAG 10
#define HID 128
#define H2 64
#define NB 2
#define SEQ 256
#define TT (SEQ - LAG)        // 246
#define ROWS (NB * TT)        // 492
#define ROWS_PAD 512
#define DD (NV * LAG)         // 320
#define PP (NV * (NV - 1))    // 992
#define H1_LD 136             // padded bf16 row stride for h1 in LDS

typedef __attribute__((ext_vector_type(16))) __bf16 v16bf;
typedef __attribute__((ext_vector_type(8)))  float  v8f;

struct Frag32 { uint4 lo, hi; };   // 32 bytes = 16 bf16 = one WMMA A/B fragment per lane

__device__ __forceinline__ v16bf frag_cast(const Frag32& f) {
  return __builtin_bit_cast(v16bf, f);
}

__device__ __forceinline__ unsigned short f2bf(float f) {
  unsigned u = __float_as_uint(f);
  u += 0x7FFFu + ((u >> 16) & 1u);   // round-to-nearest-even
  return (unsigned short)(u >> 16);
}

__device__ __forceinline__ float gelu_exact(float x) {
  return 0.5f * x * (1.0f + erff(x * 0.70710678118654752f));
}

// ---------------------------------------------------------------------------
// Prep: build feat (f32 -> d_out), feat bf16 (padded to 512 rows), targets.
// feat[b,t,(l-1)*32+v] = x[b, 10-l+t, v];  target[row,v] = x[b, t+10, v]
// ---------------------------------------------------------------------------
__global__ void gc_prep_kernel(const float* __restrict__ x,
                               float* __restrict__ outFeat,
                               unsigned short* __restrict__ featbf,
                               float* __restrict__ target)
{
  int idx = blockIdx.x * blockDim.x + threadIdx.x;
  const int NF = ROWS_PAD * DD;
  if (idx < NF) {
    int row = idx / DD, d = idx % DD;
    unsigned short bv = 0;
    if (row < ROWS) {
      int b = row / TT, t = row % TT;
      int l1 = d >> 5;                 // l-1
      int v  = d & 31;
      int s  = t + (LAG - 1) - l1;     // time index into x
      float val = x[(b * SEQ + s) * NV + v];
      outFeat[(size_t)row * DD + d] = val;
      bv = f2bf(val);
    }
    featbf[idx] = bv;                  // rows >= 492 zero-padded
  }
  if (idx < ROWS * NV) {
    int row = idx / NV, v = idx % NV;
    int b = row / TT, t = row % TT;
    target[idx] = x[(b * SEQ + (t + LAG)) * NV + v];
  }
}

// ---------------------------------------------------------------------------
// One block per model. Blocks [0,992): restricted pair p. Blocks [992,1024):
// full model for variable v = blk-992. 8 wave32 per block.
// LDS layout (bytes):
//   W1s swizzled bf16 tiles  @      0  (10 kt x 8 nt x 1024B = 81920)
//   W2s swizzled bf16 tiles  @  81920  ( 4 kt x 4 nt x 1024B = 16384)
//   b1s f32[128]             @  98304
//   b2s f32[64]              @  98816
//   W3s f32[64]              @  99072
//   pred f32[64]             @  99328
//   h1  bf16[64][136]        @  99584  (17408)
//   extras f32[2] {b3,loss}  @ 116992
// total 117000 B -> 2 workgroups per 320KB WGP
// ---------------------------------------------------------------------------
#define SOFF_W2S  81920
#define SOFF_B1   98304
#define SOFF_B2   98816
#define SOFF_W3   99072
#define SOFF_PRED 99328
#define SOFF_H1   99584
#define SOFF_EX   116992
#define SMEM_TOTAL 117000

__global__ __launch_bounds__(256)
void gc_mlp_kernel(const unsigned short* __restrict__ featbf,
                   const float* __restrict__ target,
                   const float* __restrict__ W1f, const float* __restrict__ b1f,
                   const float* __restrict__ W2f, const float* __restrict__ b2f,
                   const float* __restrict__ W3f, const float* __restrict__ b3f,
                   const float* __restrict__ W1r, const float* __restrict__ b1r,
                   const float* __restrict__ W2r, const float* __restrict__ b2r,
                   const float* __restrict__ W3r, const float* __restrict__ b3r,
                   float* __restrict__ r_loss, float* __restrict__ full_loss)
{
  extern __shared__ char smem[];
  unsigned short* W1s  = (unsigned short*)(smem);
  unsigned short* W2s  = (unsigned short*)(smem + SOFF_W2S);
  float* b1s           = (float*)(smem + SOFF_B1);
  float* b2s           = (float*)(smem + SOFF_B2);
  float* W3s           = (float*)(smem + SOFF_W3);
  float* predls        = (float*)(smem + SOFF_PRED);
  unsigned short* h1s  = (unsigned short*)(smem + SOFF_H1);
  float* extras        = (float*)(smem + SOFF_EX);   // [0]=b3, [1]=loss acc

  const int tid = threadIdx.x;
  const int blk = blockIdx.x;
  const bool restricted = (blk < PP);

  int tgtVar, srcVar;
  const float *W1g, *b1g, *W2g, *b2g, *W3g, *b3g;
  if (restricted) {
    tgtVar = blk / (NV - 1);
    int jj = blk % (NV - 1);
    srcVar = jj + (jj >= tgtVar ? 1 : 0);
    W1g = W1r + (size_t)blk * (NV - 1) * LAG * HID;
    b1g = b1r + blk * HID;
    W2g = W2r + (size_t)blk * HID * H2;
    b2g = b2r + blk * H2;
    W3g = W3r + blk * H2;
    b3g = b3r + blk;
  } else {
    int v = blk - PP;
    tgtVar = v; srcVar = -1;
    W1g = W1f + (size_t)v * DD * HID;
    b1g = b1f + v * HID;
    W2g = W2f + (size_t)v * HID * H2;
    b2g = b2f + v * H2;
    W3g = W3f + v * H2;
    b3g = b3f + v;
  }

  // --- stage W1 into LDS, pre-swizzled to B-fragment layout, masked/scattered
  for (int i = tid; i < DD * HID; i += 256) {
    int d = i / HID, h = i % HID;
    float w = 0.f;
    if (restricted) {
      int v = d & 31, l = d >> 5;
      if (v != srcVar) {
        int rrow = l * (NV - 1) + v - (v > srcVar ? 1 : 0);
        w = W1g[rrow * HID + h];
      }
    } else {
      w = W1g[d * HID + h];
    }
    int kt = d >> 5, kl = d & 31, nt = h >> 4, nl = h & 15;
    int lane = nl + ((kl & 8) ? 16 : 0);     // ISA 16-bit B layout
    W1s[(kt * 8 + nt) * 512 + lane * 16 + (kl & 7) + ((kl & 16) ? 8 : 0)] = f2bf(w);
  }
  // --- stage W2 (swizzled the same way)
  for (int i = tid; i < HID * H2; i += 256) {
    int k = i / H2, h = i % H2;
    float w = W2g[k * H2 + h];
    int kt = k >> 5, kl = k & 31, nt = h >> 4, nl = h & 15;
    int lane = nl + ((kl & 8) ? 16 : 0);
    W2s[(kt * 4 + nt) * 512 + lane * 16 + (kl & 7) + ((kl & 16) ? 8 : 0)] = f2bf(w);
  }
  if (tid < HID) b1s[tid] = b1g[tid];
  if (tid < H2)  { b2s[tid] = b2g[tid]; W3s[tid] = W3g[tid]; }
  if (tid == 0)  { extras[0] = b3g[0]; extras[1] = 0.f; }
  __syncthreads();

  const int wave = tid >> 5;
  const int lane = tid & 31;
  const int lmod = lane & 15;
  const int lhi  = lane >> 4;   // A-frag K phase / D-row phase

  const v8f vzero = {0.f,0.f,0.f,0.f,0.f,0.f,0.f,0.f};

  for (int chunk = 0; chunk < ROWS_PAD / 64; ++chunk) {
    if (tid < 64) predls[tid] = 0.f;

    // ---- GEMM1: feat(64x320) x W1(320x128) ----
    const int mt = wave >> 1;          // M tile 0..3
    const int nh = wave & 1;           // N half: columns nh*64 .. +63
    v8f acc[4] = {vzero, vzero, vzero, vzero};
    const unsigned short* arow =
        featbf + (size_t)(chunk * 64 + mt * 16 + lmod) * DD;
    for (int kt = 0; kt < 10; ++kt) {
      int kofs = kt * 32 + (lhi ? 8 : 0);
      Frag32 af;
      af.lo = *(const uint4*)(arow + kofs);        // K .. K+7
      af.hi = *(const uint4*)(arow + kofs + 16);   // K+16 .. K+23
      v16bf av = frag_cast(af);
      #pragma unroll
      for (int t = 0; t < 4; ++t) {
        const unsigned short* bp =
            W1s + (kt * 8 + nh * 4 + t) * 512 + lane * 16;
        Frag32 bfr;
        bfr.lo = *(const uint4*)(bp);
        bfr.hi = *(const uint4*)(bp + 8);
        acc[t] = __builtin_amdgcn_wmma_f32_16x16x32_bf16(
            false, av, false, frag_cast(bfr), (short)0, acc[t], false, false);
      }
    }
    __syncthreads();
    // bias + GELU -> bf16 h1 in LDS (row-major, padded stride)
    #pragma unroll
    for (int t = 0; t < 4; ++t) {
      int n = (nh * 4 + t) * 16 + lmod;
      float bias = b1s[n];
      #pragma unroll
      for (int r = 0; r < 8; ++r) {
        int m = mt * 16 + (lhi ? 8 + r : r);
        h1s[m * H1_LD + n] = f2bf(gelu_exact(acc[t][r] + bias));
      }
    }
    __syncthreads();

    // ---- GEMM2: h1(64x128) x W2(128x64) ----
    const int mt2 = wave >> 1;
    const int ntb = (wave & 1) * 2;
    v8f acc2[2] = {vzero, vzero};
    const unsigned short* arow2 = h1s + (mt2 * 16 + lmod) * H1_LD;
    for (int kt = 0; kt < 4; ++kt) {
      int kofs = kt * 32 + (lhi ? 8 : 0);
      Frag32 af;
      af.lo = *(const uint4*)(arow2 + kofs);
      af.hi = *(const uint4*)(arow2 + kofs + 16);
      v16bf av = frag_cast(af);
      #pragma unroll
      for (int t = 0; t < 2; ++t) {
        const unsigned short* bp =
            W2s + (kt * 4 + ntb + t) * 512 + lane * 16;
        Frag32 bfr;
        bfr.lo = *(const uint4*)(bp);
        bfr.hi = *(const uint4*)(bp + 8);
        acc2[t] = __builtin_amdgcn_wmma_f32_16x16x32_bf16(
            false, av, false, frag_cast(bfr), (short)0, acc2[t], false, false);
      }
    }
    // ---- layer 3: pred[m] += gelu(h2[m,n]) * W3[n] (LDS f32 atomic reduce)
    #pragma unroll
    for (int t = 0; t < 2; ++t) {
      int n = (ntb + t) * 16 + lmod;
      float bias = b2s[n], w3 = W3s[n];
      #pragma unroll
      for (int r = 0; r < 8; ++r) {
        int m = mt2 * 16 + (lhi ? 8 + r : r);
        atomicAdd(&predls[m], gelu_exact(acc2[t][r] + bias) * w3);
      }
    }
    __syncthreads();

    // ---- loss accumulation for valid rows
    if (tid < 64) {
      int row = chunk * 64 + tid;
      if (row < ROWS) {
        float d = predls[tid] + extras[0] - target[row * NV + tgtVar];
        atomicAdd(&extras[1], d * d);
      }
    }
    __syncthreads();
  }

  if (tid == 0) {
    float L = extras[1] * (1.0f / (float)ROWS);
    if (restricted) r_loss[blk] = L;
    else            full_loss[blk - PP] = L;
  }
}

// ---------------------------------------------------------------------------
// causality_matrix[src, tgt] = log(r_loss[p] / (full_loss[tgt] + 1e-8))
// ---------------------------------------------------------------------------
__global__ void gc_finalize_kernel(const float* __restrict__ r_loss,
                                   const float* __restrict__ full_loss,
                                   float* __restrict__ outMat)
{
  int tid = blockIdx.x * blockDim.x + threadIdx.x;
  if (tid < PP) {
    int i  = tid / (NV - 1);
    int jj = tid % (NV - 1);
    int j  = jj + (jj >= i ? 1 : 0);
    outMat[j * NV + i] = logf(r_loss[tid] / (full_loss[i] + 1e-8f));
  }
  if (tid < NV) outMat[tid * NV + tid] = 0.f;
}

extern "C" void kernel_launch(void* const* d_in, const int* in_sizes, int n_in,
                              void* d_out, int out_size, void* d_ws, size_t ws_size,
                              hipStream_t stream) {
  (void)in_sizes; (void)n_in; (void)out_size; (void)ws_size;
  const float* x   = (const float*)d_in[0];
  const float* W1f = (const float*)d_in[1];
  const float* b1f = (const float*)d_in[2];
  const float* W2f = (const float*)d_in[3];
  const float* b2f = (const float*)d_in[4];
  const float* W3f = (const float*)d_in[5];
  const float* b3f = (const float*)d_in[6];
  const float* W1r = (const float*)d_in[7];
  const float* b1r = (const float*)d_in[8];
  const float* W2r = (const float*)d_in[9];
  const float* b2r = (const float*)d_in[10];
  const float* W3r = (const float*)d_in[11];
  const float* b3r = (const float*)d_in[12];

  float* outMat  = (float*)d_out;            // 32x32 causality matrix
  float* outFeat = outMat + NV * NV;         // feat (2,246,320)

  // workspace layout
  char* ws = (char*)d_ws;
  unsigned short* featbf = (unsigned short*)ws;                    // 512*320*2 = 327680 B
  float* target    = (float*)(ws + 327680);                        // 492*32*4  =  62976 B
  float* full_loss = (float*)(ws + 327680 + 62976);                // 128 B
  float* r_loss    = (float*)(ws + 327680 + 62976 + 128);          // 3968 B

  gc_prep_kernel<<<(ROWS_PAD * DD + 255) / 256, 256, 0, stream>>>(
      x, outFeat, featbf, target);

  gc_mlp_kernel<<<PP + NV, 256, SMEM_TOTAL, stream>>>(
      featbf, target,
      W1f, b1f, W2f, b2f, W3f, b3f,
      W1r, b1r, W2r, b2r, W3r, b3r,
      r_loss, full_loss);

  gc_finalize_kernel<<<4, 256, 0, stream>>>(r_loss, full_loss, outMat);
}
